// Model_20486994002072
// MI455X (gfx1250) — matrix operand order repaired and checked by reading, not compiled, not run on hardware
//
#include <hip/hip_runtime.h>
#include <hip/hip_bf16.h>

typedef __attribute__((ext_vector_type(16))) _Float16 v16h;
typedef __attribute__((ext_vector_type(8)))  _Float16 v8h;
typedef __attribute__((ext_vector_type(4)))  _Float16 v4h;
typedef __attribute__((ext_vector_type(8)))  float    v8f;
typedef __attribute__((ext_vector_type(4)))  float    v4f;

#define NEG_SLOPE 0.01f

// ---------------------------------------------------------------------------
// WMMA fragment layouts, per CDNA5 ISA 7.12.2 (wave32):
//  A 16x32 f16:  lane L -> row M = L&15; 16 halfs = two contiguous 8-half
//                chunks at K = (L>>4)*8 and 16+(L>>4)*8  -> two b128 loads.
//  B 32x16 f16:  lane L -> col N = L&15; element idx i -> K = (i>>3)*16 + (L>>4)*8 + (i&7).
//                Pre-packed so each lane reads 32 contiguous bytes.
//  C/D 16x16 f32: VGPR i -> row i (lanes 0-15) / i+8 (lanes 16-31),
//                 col = lane&15.
// ---------------------------------------------------------------------------
__device__ __forceinline__ v16h cat8(v8h lo, v8h hi) {
  return __builtin_shufflevector(lo, hi, 0, 1, 2, 3, 4, 5, 6, 7,
                                 8, 9, 10, 11, 12, 13, 14, 15);
}

__device__ __forceinline__ v16h load_a_f16(const _Float16* __restrict__ A, int ld,
                                           int row0, int k0, int lane) {
  int m = lane & 15;
  int hi = lane >> 4;
  const _Float16* rp = A + (size_t)(row0 + m) * ld + k0 + hi * 8;
  v8h lo8 = *(const v8h*)rp;         // K = k0+hi*8 .. +7   (16B aligned)
  v8h hi8 = *(const v8h*)(rp + 16);  // K = k0+16+hi*8 .. +7
  return cat8(lo8, hi8);
}

__device__ __forceinline__ v16h load_a_f32(const float* __restrict__ A, int ld,
                                           int row0, int k0, int lane) {
  int m = lane & 15;
  int hi = lane >> 4;
  const float* rp = A + (size_t)(row0 + m) * ld + k0 + hi * 8;
  v4f a0 = *(const v4f*)rp;
  v4f a1 = *(const v4f*)(rp + 4);
  v4f a2 = *(const v4f*)(rp + 16);
  v4f a3 = *(const v4f*)(rp + 20);
  v16h a;
#pragma unroll
  for (int j = 0; j < 4; ++j) {
    a[j]      = (_Float16)a0[j];
    a[4 + j]  = (_Float16)a1[j];
    a[8 + j]  = (_Float16)a2[j];
    a[12 + j] = (_Float16)a3[j];
  }
  return a;
}

// Packed-B fragment: frag f, lane L -> 16 contiguous halfs at ((f*32)+L)*16.
__device__ __forceinline__ v16h load_b_packed(const _Float16* __restrict__ Bp,
                                              int frag, int lane) {
  const _Float16* p = Bp + ((size_t)frag * 32 + lane) * 16;
  v8h lo8 = *(const v8h*)p;
  v8h hi8 = *(const v8h*)(p + 8);
  return cat8(lo8, hi8);
}

// Pack fp32 weight B[K, Ncols] -> fp16 fragments [colTile][kStep][lane][16].
// Output element o = ((colTile*kSteps + kstep)*32 + lane)*16 + idx maps to
// K = kstep*32 + (idx>>3)*16 + (lane>>4)*8 + (idx&7), col = colTile*16 + (lane&15).
__global__ void pack_b_f32(const float* __restrict__ B, _Float16* __restrict__ Bp,
                           int ldN, int kSteps) {
  size_t tid = (size_t)blockIdx.x * blockDim.x + threadIdx.x;
  size_t total = (size_t)kSteps * 32 * ldN;  // K * Ncols halfs
  if (tid >= total) return;
  int idx = (int)(tid & 15);
  int lane = (int)((tid >> 4) & 31);
  size_t f = tid >> 9;
  int kstep = (int)(f % kSteps);
  int colTile = (int)(f / kSteps);
  int k = kstep * 32 + (idx >> 3) * 16 + (lane >> 4) * 8 + (idx & 7);
  int col = colTile * 16 + (lane & 15);
  Bp[tid] = (_Float16)B[(size_t)k * ldN + col];
}

// ---------------------------------------------------------------------------
// GEMM: C[M,512] (f16) = A[M,128] (f16) @ Bp(packed from [128,512]).
// Block = 256 threads = 8 waves; block owns 16 rows; wave owns 4 col tiles.
// ---------------------------------------------------------------------------
__global__ void gemm_k128_f16out(const _Float16* __restrict__ A,
                                 const _Float16* __restrict__ Bp,
                                 _Float16* __restrict__ C, int M) {
  int lane = threadIdx.x & 31;
  int wave = threadIdx.x >> 5;
  int row0 = blockIdx.x << 4;
  if (row0 + 16 > M) return;  // uniform per block: EXEC stays all-ones

  v16h afr[4];
#pragma unroll
  for (int ks = 0; ks < 4; ++ks) afr[ks] = load_a_f16(A, 128, row0, ks * 32, lane);

  int hi = lane >> 4;
  int cc = lane & 15;
#pragma unroll
  for (int t = 0; t < 4; ++t) {
    int colTile = wave * 4 + t;
    v8f acc = {};
#pragma unroll
    for (int ks = 0; ks < 4; ++ks) {
      v16h bfr = load_b_packed(Bp, colTile * 4 + ks, lane);
      acc = __builtin_amdgcn_wmma_f32_16x16x32_f16(false, afr[ks], false, bfr,
                                                   (short)0, acc, false, false);
    }
    _Float16* cp = C + (size_t)row0 * 512 + colTile * 16 + cc;
#pragma unroll
    for (int i = 0; i < 8; ++i)
      cp[(size_t)(i + hi * 8) * 512] = (_Float16)acc[i];
  }
}

// ---------------------------------------------------------------------------
// Projection GEMM: out[M,128] (f32) = A[M,512] @ Bp(packed from [512,128])
//                  + bias + resid.  8 waves x one 16-col tile = 128 cols.
// ---------------------------------------------------------------------------
template <bool A_IS_HALF>
__global__ void gemm_k512_proj(const void* __restrict__ Av,
                               const _Float16* __restrict__ Bp,
                               const float* __restrict__ bias,
                               const float* __restrict__ resid,
                               float* __restrict__ out, int M) {
  int lane = threadIdx.x & 31;
  int wave = threadIdx.x >> 5;
  int row0 = blockIdx.x << 4;
  if (row0 + 16 > M) return;

  // Prefetch the epilogue residual rows while the WMMA K-loop runs
  // (lowers to global_prefetch_b8; each lane pulls one row's cacheline).
  {
    int col = wave * 16 + (lane & 15);
    int r = row0 + (lane >> 4) * 8 + (lane & 7);
    __builtin_prefetch(resid + (size_t)r * 128 + col, 0, 3);
  }

  v8f acc = {};
#pragma unroll
  for (int ks = 0; ks < 16; ++ks) {
    v16h afr = A_IS_HALF ? load_a_f16((const _Float16*)Av, 512, row0, ks * 32, lane)
                         : load_a_f32((const float*)Av, 512, row0, ks * 32, lane);
    v16h bfr = load_b_packed(Bp, wave * 16 + ks, lane);
    acc = __builtin_amdgcn_wmma_f32_16x16x32_f16(false, afr, false, bfr,
                                                 (short)0, acc, false, false);
  }
  int hi = lane >> 4;
  int col = wave * 16 + (lane & 15);
  float bv = bias[col];
#pragma unroll
  for (int i = 0; i < 8; ++i) {
    size_t r = (size_t)(row0 + i + hi * 8);
    out[r * 128 + col] = acc[i] + bv + resid[r * 128 + col];
  }
}

// ---------------------------------------------------------------------------
// Helpers / elementwise kernels
// ---------------------------------------------------------------------------
__device__ __forceinline__ void atomicMaxF(float* addr, float val) {
  if (val >= 0.0f)
    atomicMax((int*)addr, __float_as_int(val));
  else
    atomicMin((unsigned int*)addr, (unsigned int)__float_as_int(val));
}

// 8 elements per thread (n must be a multiple of 8; all our sizes are).
__global__ void cvt_f32_to_f16_x8(const float* __restrict__ in,
                                  _Float16* __restrict__ out, size_t n8) {
  size_t tid = (size_t)blockIdx.x * blockDim.x + threadIdx.x;
  if (tid >= n8) return;
  const v4f* ip = (const v4f*)(in + tid * 8);
  v4f a = ip[0], b = ip[1];
  v8h o;
#pragma unroll
  for (int j = 0; j < 4; ++j) {
    o[j] = (_Float16)a[j];
    o[4 + j] = (_Float16)b[j];
  }
  *(v8h*)(out + tid * 8) = o;
}

__global__ void init_accum(float* __restrict__ nh_acc, float* __restrict__ mbuf,
                           float* __restrict__ denom, int N) {
  int tid = blockIdx.x * blockDim.x + threadIdx.x;
  if (tid < N * 512) nh_acc[tid] = 0.0f;
  if (tid < N * 4) {
    mbuf[tid] = -__builtin_inff();
    denom[tid] = 0.0f;
  }
}

// Wave-per-edge: f_out (in place over f_fij), attention logits e, seg-max.
__global__ void edge_attn_kernel(_Float16* __restrict__ fe,         // [E,512]
                                 const _Float16* __restrict__ fni,  // [N,512]
                                 const _Float16* __restrict__ fnj,  // [N,512]
                                 const int* __restrict__ src,
                                 const int* __restrict__ dst,
                                 const float* __restrict__ bias,    // [512]
                                 const float* __restrict__ attn,    // [512]
                                 float* __restrict__ ebuf,          // [E*4]
                                 float* __restrict__ mbuf,          // [N*4]
                                 int E) {
  int lane = threadIdx.x & 31;
  int wave = threadIdx.x >> 5;
  int e = blockIdx.x * 8 + wave;
  if (e >= E) return;
  int s = src[e], d = dst[e];
  const _Float16* pni = fni + (size_t)s * 512;
  const _Float16* pnj = fnj + (size_t)d * 512;
  _Float16* pe = fe + (size_t)e * 512;
  // Prefetch the gathered node rows (random-access) up front.
  __builtin_prefetch(pni + lane * 16, 0, 2);
  __builtin_prefetch(pnj + lane * 16, 0, 2);
#pragma unroll
  for (int h = 0; h < 4; ++h) {
    int cbase = h * 128 + lane * 4;
    v4h fij = *(const v4h*)(pe + cbase);
    v4h ni  = *(const v4h*)(pni + cbase);
    v4h nj  = *(const v4h*)(pnj + cbase);
    v4f bv  = *(const v4f*)(bias + cbase);
    v4f av  = *(const v4f*)(attn + cbase);
    float psum = 0.0f;
    v4h ov;
#pragma unroll
    for (int j = 0; j < 4; ++j) {
      float v = (float)fij[j] + (float)ni[j] + (float)nj[j] + bv[j];
      v = v >= 0.0f ? v : NEG_SLOPE * v;
      ov[j] = (_Float16)v;
      psum += v * av[j];
    }
    *(v4h*)(pe + cbase) = ov;
#pragma unroll
    for (int off = 16; off > 0; off >>= 1) psum += __shfl_xor(psum, off, 32);
    if (lane == 0) {
      ebuf[(size_t)e * 4 + h] = psum;
      atomicMaxF(&mbuf[(size_t)d * 4 + h], psum);
    }
  }
}

__global__ void softmax_phase2(float* __restrict__ ebuf,
                               const float* __restrict__ mbuf,
                               float* __restrict__ denom,
                               const int* __restrict__ dst, int E) {
  int tid = blockIdx.x * blockDim.x + threadIdx.x;
  if (tid >= E * 4) return;
  int e = tid >> 2, h = tid & 3;
  int d = dst[e];
  float ex = __expf(ebuf[tid] - mbuf[(size_t)d * 4 + h]);
  ebuf[tid] = ex;  // overwrite logit with exp (own index only)
  atomicAdd(&denom[(size_t)d * 4 + h], ex);
}

// Wave-per-edge scatter: nh_acc[dst] += a * h_src[src]
__global__ void scatter_kernel(const float* __restrict__ ebuf,
                               const float* __restrict__ denom,
                               const _Float16* __restrict__ hsrc,  // [N,512]
                               const int* __restrict__ src,
                               const int* __restrict__ dst,
                               float* __restrict__ nh_acc,         // [N,512]
                               int E) {
  int lane = threadIdx.x & 31;
  int wave = threadIdx.x >> 5;
  int e = blockIdx.x * 8 + wave;
  if (e >= E) return;
  int s = src[e], d = dst[e];
  const _Float16* ps = hsrc + (size_t)s * 512;
  float* pd = nh_acc + (size_t)d * 512;
  __builtin_prefetch(ps + lane * 16, 0, 2);
#pragma unroll
  for (int h = 0; h < 4; ++h) {
    float a = ebuf[(size_t)e * 4 + h] / denom[(size_t)d * 4 + h];
    int cbase = h * 128 + lane * 4;
    v4h hv = *(const v4h*)(ps + cbase);
#pragma unroll
    for (int j = 0; j < 4; ++j)
      atomicAdd(&pd[cbase + j], (float)hv[j] * a);
  }
}

// ---------------------------------------------------------------------------
extern "C" void kernel_launch(void* const* d_in, const int* in_sizes, int n_in,
                              void* d_out, int out_size, void* d_ws, size_t ws_size,
                              hipStream_t stream) {
  const float* nfeats   = (const float*)d_in[0];
  const float* efeats   = (const float*)d_in[1];
  const int*   src      = (const int*)d_in[2];
  const int*   dst      = (const int*)d_in[3];
  const float* W_nsrc   = (const float*)d_in[4];
  const float* W_ni     = (const float*)d_in[5];
  const float* W_nj     = (const float*)d_in[6];
  const float* W_fij    = (const float*)d_in[7];
  const float* attn     = (const float*)d_in[8];
  const float* b_egat   = (const float*)d_in[9];
  const float* W_proj_n = (const float*)d_in[10];
  const float* b_proj_n = (const float*)d_in[11];
  const float* W_proj_e = (const float*)d_in[12];
  const float* b_proj_e = (const float*)d_in[13];

  const int D = 128, HD = 512;
  const int N = in_sizes[0] / D;
  const int E = in_sizes[2];

  // ---- workspace layout (256B aligned slices) ----
  char* ws = (char*)d_ws;
  auto alloc = [&](size_t bytes) -> char* {
    char* p = ws;
    ws += (bytes + 255) & ~(size_t)255;
    return p;
  };
  _Float16* nf16   = (_Float16*)alloc((size_t)N * D * 2);
  _Float16* ef16   = (_Float16*)alloc((size_t)E * D * 2);
  _Float16* Wni_p  = (_Float16*)alloc((size_t)D * HD * 2);  // packed fragments
  _Float16* Wnj_p  = (_Float16*)alloc((size_t)D * HD * 2);
  _Float16* Wsrc_p = (_Float16*)alloc((size_t)D * HD * 2);
  _Float16* Wfij_p = (_Float16*)alloc((size_t)D * HD * 2);
  _Float16* Wpn_p  = (_Float16*)alloc((size_t)HD * D * 2);
  _Float16* Wpe_p  = (_Float16*)alloc((size_t)HD * D * 2);
  _Float16* fni16  = (_Float16*)alloc((size_t)N * HD * 2);
  _Float16* fnj16  = (_Float16*)alloc((size_t)N * HD * 2);
  _Float16* hsrc16 = (_Float16*)alloc((size_t)N * HD * 2);
  _Float16* fout16 = (_Float16*)alloc((size_t)E * HD * 2);  // f_fij -> f_out
  float*    ebuf   = (float*)alloc((size_t)E * 4 * 4);
  float*    mbuf   = (float*)alloc((size_t)N * 4 * 4);
  float*    denom  = (float*)alloc((size_t)N * 4 * 4);
  float*    nh_acc = (float*)alloc((size_t)N * HD * 4);

  float* out_nh = (float*)d_out;
  float* out_eh = out_nh + (size_t)N * D;

  const int T = 256;
  auto cdiv = [](size_t a, size_t b) { return (unsigned)((a + b - 1) / b); };

  // 1) fp32 -> fp16 feature conversions (8 elems/thread)
  cvt_f32_to_f16_x8<<<cdiv((size_t)N * D / 8, T), T, 0, stream>>>(nfeats, nf16,
                                                                  (size_t)N * D / 8);
  cvt_f32_to_f16_x8<<<cdiv((size_t)E * D / 8, T), T, 0, stream>>>(efeats, ef16,
                                                                  (size_t)E * D / 8);

  // 2) pack weights into WMMA-B fragment layout (fuses fp32->fp16)
  //    K=128, Ncols=512 -> kSteps=4; K=512, Ncols=128 -> kSteps=16.
  pack_b_f32<<<cdiv((size_t)D * HD, T), T, 0, stream>>>(W_ni, Wni_p, HD, 4);
  pack_b_f32<<<cdiv((size_t)D * HD, T), T, 0, stream>>>(W_nj, Wnj_p, HD, 4);
  pack_b_f32<<<cdiv((size_t)D * HD, T), T, 0, stream>>>(W_nsrc, Wsrc_p, HD, 4);
  pack_b_f32<<<cdiv((size_t)D * HD, T), T, 0, stream>>>(W_fij, Wfij_p, HD, 4);
  pack_b_f32<<<cdiv((size_t)HD * D, T), T, 0, stream>>>(W_proj_n, Wpn_p, D, 16);
  pack_b_f32<<<cdiv((size_t)HD * D, T), T, 0, stream>>>(W_proj_e, Wpe_p, D, 16);

  // 3) init accumulators (every call; harness does not re-poison)
  init_accum<<<cdiv((size_t)N * HD, T), T, 0, stream>>>(nh_acc, mbuf, denom, N);

  // 4) WMMA GEMMs, K=128
  gemm_k128_f16out<<<cdiv(N, 16), T, 0, stream>>>(nf16, Wni_p, fni16, N);
  gemm_k128_f16out<<<cdiv(N, 16), T, 0, stream>>>(nf16, Wnj_p, fnj16, N);
  gemm_k128_f16out<<<cdiv(N, 16), T, 0, stream>>>(nf16, Wsrc_p, hsrc16, N);
  gemm_k128_f16out<<<cdiv(E, 16), T, 0, stream>>>(ef16, Wfij_p, fout16, E);

  // 5) edge combine + leaky-relu + attention logits + segment max
  edge_attn_kernel<<<cdiv(E, 8), T, 0, stream>>>(fout16, fni16, fnj16, src, dst,
                                                 b_egat, attn, ebuf, mbuf, E);

  // 6) edge softmax: exp + segment sum
  softmax_phase2<<<cdiv((size_t)E * 4, T), T, 0, stream>>>(ebuf, mbuf, denom, dst, E);

  // 7) weighted message scatter into nh_acc
  scatter_kernel<<<cdiv(E, 8), T, 0, stream>>>(ebuf, denom, hsrc16, src, dst, nh_acc, E);

  // 8) projections + bias + residual -> d_out (nh then eh)
  gemm_k512_proj<false><<<cdiv(N, 16), T, 0, stream>>>(nh_acc, Wpn_p, b_proj_n,
                                                       nfeats, out_nh, N);
  gemm_k512_proj<true><<<cdiv(E, 16), T, 0, stream>>>(fout16, Wpe_p, b_proj_e,
                                                      efeats, out_eh, E);
}
